// SelfAttention_9895604650459
// MI455X (gfx1250) — matrix operand order repaired and checked by reading, not compiled, not run on hardware
//
#include <hip/hip_runtime.h>
#include <hip/hip_bf16.h>

// ---------------------------------------------------------------------------
// Causal MHSA, B=4 N=2048 D=1024 H=16 HD=64, f32 in/out, f16 WMMA compute.
// Software-pipelined: operand loads for step i+1 issue before WMMAs of step i.
// ---------------------------------------------------------------------------

typedef _Float16 v16h __attribute__((ext_vector_type(16)));
typedef _Float16 v8h  __attribute__((ext_vector_type(8)));
typedef float    v8f  __attribute__((ext_vector_type(8)));

#define WMMA16(a, b, c) \
    __builtin_amdgcn_wmma_f32_16x16x32_f16(false, (a), false, (b), (short)0, (c), false, false)

static __device__ __forceinline__ v16h cat16(v8h lo, v8h hi) {
    return __builtin_shufflevector(lo, hi, 0,1,2,3,4,5,6,7,8,9,10,11,12,13,14,15);
}

#define SWZ(v, imm) __int_as_float(__builtin_amdgcn_ds_swizzle(__float_as_int(v), (imm)))
static __device__ __forceinline__ float redmax16(float t) {
    // max over the 16-lane half-group (XOR butterfly, stays inside lanes 0-15 / 16-31)
    t = fmaxf(t, SWZ(t, 0x041F));
    t = fmaxf(t, SWZ(t, 0x081F));
    t = fmaxf(t, SWZ(t, 0x101F));
    t = fmaxf(t, SWZ(t, 0x201F));
    return t;
}
static __device__ __forceinline__ float redsum16(float t) {
    t += SWZ(t, 0x041F);
    t += SWZ(t, 0x081F);
    t += SWZ(t, 0x101F);
    t += SWZ(t, 0x201F);
    return t;
}

#if __has_builtin(__builtin_amdgcn_exp2f)
#define EXP2F __builtin_amdgcn_exp2f
#else
#define EXP2F exp2f
#endif
#define LOG2E 1.4426950408889634f

static constexpr int Bc = 4, Nc = 2048, Dc = 1024, Hc = 16, HDc = 64;

// ---------------------------------------------------------------------------
// Kernel 1: f32 -> f16 cast of x
// ---------------------------------------------------------------------------
__global__ void k_cvt_x(const float* __restrict__ x, _Float16* __restrict__ xh, int n) {
    int i = blockIdx.x * blockDim.x + threadIdx.x;
    if (i < n) xh[i] = (_Float16)x[i];
}

// ---------------------------------------------------------------------------
// Kernel 2: transpose+cast weights: Wt[e*1024+d] = W[d*1024+e]   (B^T layout)
// ---------------------------------------------------------------------------
__global__ void k_cvt_w(const float* __restrict__ wq, const float* __restrict__ wk,
                        const float* __restrict__ wv, _Float16* __restrict__ tq,
                        _Float16* __restrict__ tk, _Float16* __restrict__ tv) {
    int mat = blockIdx.y;
    const float* src = (mat == 0) ? wq : (mat == 1) ? wk : wv;
    _Float16*    dst = (mat == 0) ? tq : (mat == 1) ? tk : tv;
    int i = blockIdx.x * 256 + threadIdx.x;     // 0 .. 1M-1
    int d = i >> 10;
    int e = i & 1023;
    dst[e * 1024 + d] = (_Float16)src[d * 1024 + e];
}

// ---------------------------------------------------------------------------
// Kernel 3: QKV projection GEMM.  Each wave: one 16(M) x 64(N) tile,
// double-buffered over the K dimension.
//   Q -> Qh[b,h,n,d] f16, pre-scaled by 1/8
//   K -> Kh[b,h,n,d] f16                 (== B operand of Q*K^T)
//   V -> Vt[b,h,d,n] f16 (transposed)    (== B operand of P*V)
// ---------------------------------------------------------------------------
__global__ __launch_bounds__(256) void
k_qkv(const _Float16* __restrict__ xh,
      const _Float16* __restrict__ wqt, const _Float16* __restrict__ wkt,
      const _Float16* __restrict__ wvt,
      const float* __restrict__ bq, const float* __restrict__ bk,
      const float* __restrict__ bv,
      _Float16* __restrict__ qh, _Float16* __restrict__ kh,
      _Float16* __restrict__ vt) {
    const int wid  = blockIdx.x * 8 + (threadIdx.x >> 5);
    const int lane = threadIdx.x & 31;
    const int lm   = lane & 15;
    const int hi   = lane >> 4;

    const int mat = wid >> 13;          // 8192 wave-tiles per matrix
    const int r   = wid & 8191;
    const int m0  = (r >> 4) << 4;      // 512 M-tiles of 16
    const int n0  = (r & 15) << 6;      // 16 N-tiles of 64

    const _Float16* Wt   = (mat == 0) ? wqt : (mat == 1) ? wkt : wvt;
    const float*    bias = (mat == 0) ? bq  : (mat == 1) ? bk  : bv;

    v8f acc[4];
#pragma unroll
    for (int t = 0; t < 4; ++t) acc[t] = (v8f){};

    const _Float16* arow = xh + (size_t)(m0 + lm) * Dc;
    const _Float16* brow = Wt + (size_t)(n0 + lm) * Dc;

    auto ldA = [&](int k) -> v16h {
        return cat16(*(const v8h*)(arow + k + hi * 8),
                     *(const v8h*)(arow + k + 16 + hi * 8));
    };
    auto ldB = [&](int t, int k) -> v16h {
        const _Float16* bp = brow + (size_t)t * 16 * Dc;
        return cat16(*(const v8h*)(bp + k + hi * 8),
                     *(const v8h*)(bp + k + 16 + hi * 8));
    };

    // prologue: buffer for k = 0
    v16h a  = ldA(0);
    v16h b0 = ldB(0, 0), b1 = ldB(1, 0), b2 = ldB(2, 0), b3 = ldB(3, 0);

    for (int k = 0; k < Dc; k += 32) {
        // issue next step's loads first (clamped on last iteration: harmless reload)
        const int kn = (k + 32 < Dc) ? (k + 32) : k;
        v16h aN  = ldA(kn);
        v16h b0N = ldB(0, kn), b1N = ldB(1, kn), b2N = ldB(2, kn), b3N = ldB(3, kn);

        acc[0] = WMMA16(a, b0, acc[0]);
        acc[1] = WMMA16(a, b1, acc[1]);
        acc[2] = WMMA16(a, b2, acc[2]);
        acc[3] = WMMA16(a, b3, acc[3]);

        a = aN; b0 = b0N; b1 = b1N; b2 = b2N; b3 = b3N;
    }

#pragma unroll
    for (int t = 0; t < 4; ++t) {
        int col = n0 + t * 16 + lm;
        float bv_ = bias[col];
        int h = col >> 6, d = col & 63;
#pragma unroll
        for (int j = 0; j < 8; ++j) {
            int row = m0 + j + 8 * hi;           // global M row (= b*2048+n)
            int b   = row >> 11, n = row & 2047;
            float v = acc[t][j] + bv_;
            if (mat == 0) {
                qh[((size_t)(b * Hc + h) * Nc + n) * HDc + d] = (_Float16)(v * 0.125f);
            } else if (mat == 1) {
                kh[((size_t)(b * Hc + h) * Nc + n) * HDc + d] = (_Float16)v;
            } else {
                vt[((size_t)(b * Hc + h) * HDc + d) * Nc + n] = (_Float16)v;
            }
        }
    }
}

// ---------------------------------------------------------------------------
// Kernel 4: flash attention.  Grid (B*H, N/64), 128 threads = 4 waves.
// Each wave: 16 query rows, online softmax over key blocks of 32.
// Pipelined: V tiles + next K tiles are issued before the softmax VALU phase.
// ---------------------------------------------------------------------------
struct KT { v16h c0, c1; };

__global__ __launch_bounds__(128) void
k_attn(const _Float16* __restrict__ qh, const _Float16* __restrict__ kh,
       const _Float16* __restrict__ vt, float* __restrict__ out) {
    __shared__ _Float16 pbuf[4][16 * 32];   // per-wave P-transpose staging (1 KB each)

    const int bh   = blockIdx.x;            // b*16 + h
    const int wave = threadIdx.x >> 5;
    const int lane = threadIdx.x & 31;
    const int lm   = lane & 15;
    const int hi   = lane >> 4;
    const int q0   = blockIdx.y * 64 + wave * 16;
    const int qmax = q0 + 15;

    // Q A-operands (held for the whole loop), already scaled by 1/8
    const _Float16* qb = qh + ((size_t)bh * Nc + q0 + lm) * HDc;
    v16h qa0 = cat16(*(const v8h*)(qb + hi * 8),      *(const v8h*)(qb + 16 + hi * 8));
    v16h qa1 = cat16(*(const v8h*)(qb + 32 + hi * 8), *(const v8h*)(qb + 48 + hi * 8));

    auto ldK = [&](int keybase) -> KT {
        const _Float16* kb = kh + ((size_t)bh * Nc + keybase + lm) * HDc;
        KT r;
        r.c0 = cat16(*(const v8h*)(kb + hi * 8),      *(const v8h*)(kb + 16 + hi * 8));
        r.c1 = cat16(*(const v8h*)(kb + 32 + hi * 8), *(const v8h*)(kb + 48 + hi * 8));
        return r;
    };
    auto ldV = [&](int t, int kb) -> v16h {
        const _Float16* vb = vt + ((size_t)bh * HDc + t * 16 + lm) * Nc + kb;
        return cat16(*(const v8h*)(vb + hi * 8), *(const v8h*)(vb + 16 + hi * 8));
    };

    v8f o[4];
#pragma unroll
    for (int t = 0; t < 4; ++t) o[t] = (v8f){};
    float mrow[8], srow[8];
#pragma unroll
    for (int j = 0; j < 8; ++j) { mrow[j] = -3.0e38f; srow[j] = 0.0f; }

    _Float16* pw = &pbuf[wave][0];

    // prologue: K tiles for the first key block
    KT kt0 = ldK(0), kt1 = ldK(16);

    for (int k0 = 0; k0 <= qmax; k0 += 32) {
        // ---- scores: two 16x16 tiles (keys k0..k0+15 and k0+16..k0+31) ----
        v8f S0 = WMMA16(qa1, kt0.c1, WMMA16(qa0, kt0.c0, (v8f){}));
        v8f S1 = WMMA16(qa1, kt1.c1, WMMA16(qa0, kt1.c0, (v8f){}));

        // ---- issue V tiles for this block + K tiles for the next block;
        //      their latency hides behind the softmax VALU phase ----
        v16h vb0 = ldV(0, k0), vb1 = ldV(1, k0), vb2 = ldV(2, k0), vb3 = ldV(3, k0);
        const int k0n = (k0 + 32 <= qmax) ? (k0 + 32) : k0;
        KT kt0N = ldK(k0n), kt1N = ldK(k0n + 16);

        // ---- causal mask (only diagonal blocks; wave-uniform branch) ----
        if (k0 + 31 > q0) {
            int c0 = k0 + lm, c1 = c0 + 16;
#pragma unroll
            for (int j = 0; j < 8; ++j) {
                int rowj = q0 + j + 8 * hi;
                if (c0 > rowj) S0[j] = -1.0e30f;
                if (c1 > rowj) S1[j] = -1.0e30f;
            }
        }

        // ---- online softmax update ----
        float sc[8];
#pragma unroll
        for (int j = 0; j < 8; ++j) {
            float rmax = redmax16(fmaxf(S0[j], S1[j]));
            float nm   = fmaxf(mrow[j], rmax);
            sc[j]      = EXP2F((mrow[j] - nm) * LOG2E);
            mrow[j]    = nm;
            float p0   = EXP2F((S0[j] - nm) * LOG2E);
            float p1   = EXP2F((S1[j] - nm) * LOG2E);
            S0[j] = p0; S1[j] = p1;
            srow[j] = srow[j] * sc[j] + redsum16(p0 + p1);
        }
#pragma unroll
        for (int t = 0; t < 4; ++t)
#pragma unroll
            for (int j = 0; j < 8; ++j) o[t][j] *= sc[j];

        // ---- C-layout -> A-layout transpose of P through LDS ----
#pragma unroll
        for (int j = 0; j < 8; ++j) {
            pw[(j + 8 * hi) * 32 + lm]      = (_Float16)S0[j];
            pw[(j + 8 * hi) * 32 + 16 + lm] = (_Float16)S1[j];
        }
        asm volatile("s_wait_dscnt 0x0" ::: "memory");
        const _Float16* pr = pw + lm * 32;
        v16h pa = cat16(*(const v8h*)(pr + hi * 8), *(const v8h*)(pr + 16 + hi * 8));

        // ---- P(16x32) @ V(32x64): 4 WMMAs into persistent accumulators ----
        o[0] = WMMA16(pa, vb0, o[0]);
        o[1] = WMMA16(pa, vb1, o[1]);
        o[2] = WMMA16(pa, vb2, o[2]);
        o[3] = WMMA16(pa, vb3, o[3]);

        kt0 = kt0N; kt1 = kt1N;
    }

    // ---- normalize + store f32 output [B,N,D] with head de-interleave ----
    const int b = bh >> 4, h = bh & 15;
#pragma unroll
    for (int t = 0; t < 4; ++t) {
        int col = h * HDc + t * 16 + lm;
#pragma unroll
        for (int j = 0; j < 8; ++j) {
            int row = q0 + j + 8 * hi;
            out[((size_t)(b * Nc + row)) * Dc + col] = o[t][j] / srow[j];
        }
    }
}

// ---------------------------------------------------------------------------
// Host launcher
// ---------------------------------------------------------------------------
extern "C" void kernel_launch(void* const* d_in, const int* in_sizes, int n_in,
                              void* d_out, int out_size, void* d_ws, size_t ws_size,
                              hipStream_t stream) {
    const float* x  = (const float*)d_in[0];
    // d_in[1] = mask (unused: causality is hardcoded)
    const float* Wq = (const float*)d_in[2];
    const float* bq = (const float*)d_in[3];
    const float* Wk = (const float*)d_in[4];
    const float* bk = (const float*)d_in[5];
    const float* Wv = (const float*)d_in[6];
    const float* bv = (const float*)d_in[7];
    float* out = (float*)d_out;

    char* ws = (char*)d_ws;
    const size_t MB = 1u << 20;
    _Float16* Xh  = (_Float16*)(ws);              // 16 MB  (8192 x 1024 f16)
    _Float16* Wqt = (_Float16*)(ws + 16 * MB);    //  2 MB
    _Float16* Wkt = (_Float16*)(ws + 18 * MB);    //  2 MB
    _Float16* Wvt = (_Float16*)(ws + 20 * MB);    //  2 MB
    _Float16* Qh  = (_Float16*)(ws + 22 * MB);    // 16 MB  [B,H,N,64]
    _Float16* Kh  = (_Float16*)(ws + 38 * MB);    // 16 MB  [B,H,N,64]
    _Float16* Vt  = (_Float16*)(ws + 54 * MB);    // 16 MB  [B,H,64,N]

    const int nx = Bc * Nc * Dc;                  // 8,388,608
    k_cvt_x<<<nx / 256, 256, 0, stream>>>(x, Xh, nx);
    k_cvt_w<<<dim3(4096, 3), 256, 0, stream>>>(Wq, Wk, Wv, Wqt, Wkt, Wvt);
    // 3 mats * 512 Mtiles * 16 Ntiles = 24576 waves / 8 waves per block
    k_qkv<<<3072, 256, 0, stream>>>(Xh, Wqt, Wkt, Wvt, bq, bk, bv, Qh, Kh, Vt);
    k_attn<<<dim3(Bc * Hc, Nc / 64), 128, 0, stream>>>(Qh, Kh, Vt, out);
}
